// ConvSlimCapsule3D_33818572489333
// MI455X (gfx1250) — matrix operand order repaired and checked by reading, not compiled, not run on hardware
//
#include <hip/hip_runtime.h>

typedef __attribute__((ext_vector_type(16))) _Float16 v16h;
typedef __attribute__((ext_vector_type(8)))  _Float16 v8h;
typedef __attribute__((ext_vector_type(8)))  float    v8f;

#define IN_DIM   8
#define IN_ATOMS 16
#define OUT_DIM  8
#define OUT_ATOMS 16
#define NCH      128            // OUT_DIM * OUT_ATOMS
#define KTOT     432            // IN_ATOMS * 27
#define KPAD     448            // 14 * 32
#define KSTEPS   14
#define NTILES   8              // 128 / 16
#define MT       16             // spatial positions per tile
#define SP       32768          // 32^3

// ---------------------------------------------------------------------------
// Pre-pack conv weights (128 x 432 fp32) into WMMA B-fragment layout, f16.
// Consumed as: frag base = ((ntile*KSTEPS + kstep)*32 + lane)*16 halves.
// B packing used here (32x16 KxN): lane holds N = lane%16; element e -> K = e + 16*(lane/16) (same k order as A).
// ---------------------------------------------------------------------------
__global__ void prep_weights_kernel(const float* __restrict__ w,
                                    _Float16* __restrict__ bpack) {
    int tid = blockIdx.x * 256 + threadIdx.x;
    if (tid >= NTILES * KSTEPS * 32 * 16) return;
    int e     = tid & 15;
    int lane  = (tid >> 4) & 31;
    int kstep = (tid >> 9) % KSTEPS;
    int ntile = tid / (KSTEPS * 512);
    int N = ntile * 16 + (lane & 15);
    int K = e + ((lane >> 4) << 4) + kstep * 32;
    float v = (K < KTOT) ? w[N * KTOT + K] : 0.0f;
    bpack[tid] = (_Float16)v;
}

__device__ __forceinline__ float reduce16(float x) {
    // sum across the 16-lane position group (masks < 16 never cross the group)
    x += __shfl_xor(x, 1, 32);
    x += __shfl_xor(x, 2, 32);
    x += __shfl_xor(x, 4, 32);
    x += __shfl_xor(x, 8, 32);
    return x;
}

// ---------------------------------------------------------------------------
// Fused: implicit-GEMM conv3d (WMMA f16->f32) + 3-round dynamic routing + squash
// Grid: 4096 blocks (2 batches x 2048 row-tiles of 16 positions), 256 threads.
// ---------------------------------------------------------------------------
__launch_bounds__(256, 1)
__global__ void caps_fused_kernel(const float* __restrict__ x,
                                  const _Float16* __restrict__ bpack,
                                  const float* __restrict__ conv_b,
                                  const float* __restrict__ biases,
                                  float* __restrict__ out) {
    // 8 waves * (16 x 448) f16 im2col tiles = 114688 B; reused as votes f32 (64 KB)
    __shared__ __align__(16) _Float16 lds_h[IN_DIM * MT * KPAD];
    float* vl = (float*)lds_h;   // votes: [i][m][n] : 8*16*128 f32

    const int tid  = threadIdx.x;
    const int wave = tid >> 5;
    const int lane = tid & 31;

    const int blk = blockIdx.x;
    const int b   = blk >> 11;            // 0..1
    const int t   = blk & 2047;
    const int z   = t >> 6;               // 0..31
    const int y   = (t >> 1) & 31;        // 0..31
    const int x0  = (t & 1) << 4;         // 0 or 16

    // ---- Stage A: this wave's im2col patch (own-wave LDS only, no barrier) ----
    const int i = wave;                   // in_dim index
    const float* xim = x + (size_t)(b * IN_DIM + i) * IN_ATOMS * SP;
    _Float16* As = lds_h + i * (MT * KPAD);
    for (int tt = lane; tt < MT * KPAD; tt += 32) {
        int M = tt / KPAD;
        int K = tt - M * KPAD;
        float v = 0.0f;
        if (K < KTOT) {
            int ia = K / 27;
            int r  = K - ia * 27;
            int kz = r / 9;
            int r2 = r - kz * 9;
            int ky = r2 / 3;
            int kx = r2 - ky * 3;
            int zi = z + kz - 1, yi = y + ky - 1, xx = x0 + M + kx - 1;
            if ((unsigned)zi < 32u && (unsigned)yi < 32u && (unsigned)xx < 32u)
                v = xim[ia * SP + zi * 1024 + yi * 32 + xx];
        }
        As[tt] = (_Float16)v;
    }

    // ---- Implicit GEMM: M=16 positions x N=128 channels, K=448 ----
    v8f acc[NTILES] = {};
    const int Mrow = lane & 15;
    const int ksub = (lane >> 4) << 4;    // +16 K offset for upper half-wave (contiguous k order, matches bpack)
#pragma unroll
    for (int ks = 0; ks < KSTEPS; ++ks) {
        const _Float16* ap = As + Mrow * KPAD + ks * 32 + ksub;
        v8h alo = *(const v8h*)ap;
        v8h ahi = *(const v8h*)(ap + 8);
        v16h afrag = __builtin_shufflevector(alo, ahi,
            0,1,2,3,4,5,6,7,8,9,10,11,12,13,14,15);
#pragma unroll
        for (int nt = 0; nt < NTILES; ++nt) {
            const _Float16* bp = bpack + ((size_t)((nt * KSTEPS + ks) * 32 + lane)) * 16;
            v8h blo = *(const v8h*)bp;
            v8h bhi = *(const v8h*)(bp + 8);
            v16h bfrag = __builtin_shufflevector(blo, bhi,
                0,1,2,3,4,5,6,7,8,9,10,11,12,13,14,15);
            acc[nt] = __builtin_amdgcn_wmma_f32_16x16x32_f16(
                false, afrag, false, bfrag, (short)0, acc[nt], false, false);
        }
    }

    // ---- Votes (+conv bias) to LDS; C/D layout: VGPR r -> M = r + 8*(lane/16) ----
    __syncthreads();   // all waves done reading their As before it's overwritten
    {
        const int rowhi = (lane >> 4) << 3;
        const int col   = lane & 15;
#pragma unroll
        for (int nt = 0; nt < NTILES; ++nt) {
            float cb = conv_b[nt * 16 + col];
#pragma unroll
            for (int rr = 0; rr < 8; ++rr) {
                int M = rr + rowhi;
                vl[i * (MT * NCH) + M * NCH + nt * 16 + col] = acc[nt][rr] + cb;
            }
        }
    }
    __syncthreads();

    // ---- Dynamic routing: 16 threads per position, thread owns atom a ----
    const int m = tid >> 4;               // position in tile
    const int a = tid & 15;               // out_atom

    float v[IN_DIM][OUT_DIM];
#pragma unroll
    for (int ii = 0; ii < IN_DIM; ++ii)
#pragma unroll
        for (int o = 0; o < OUT_DIM; ++o)
            v[ii][o] = vl[ii * (MT * NCH) + m * NCH + o * 16 + a];

    float bias_v[OUT_DIM];
#pragma unroll
    for (int o = 0; o < OUT_DIM; ++o) bias_v[o] = biases[o * 16 + a];

    float logits[IN_DIM][OUT_DIM] = {};
    float pre[OUT_DIM];

#pragma unroll
    for (int it = 0; it < 3; ++it) {
#pragma unroll
        for (int o = 0; o < OUT_DIM; ++o) pre[o] = bias_v[o];
        // route = softmax over out_dim; preact = sum_i votes*route + bias
#pragma unroll
        for (int ii = 0; ii < IN_DIM; ++ii) {
            float mx = logits[ii][0];
#pragma unroll
            for (int o = 1; o < OUT_DIM; ++o) mx = fmaxf(mx, logits[ii][o]);
            float ev[OUT_DIM], se = 0.0f;
#pragma unroll
            for (int o = 0; o < OUT_DIM; ++o) { ev[o] = __expf(logits[ii][o] - mx); se += ev[o]; }
            float inv = 1.0f / se;
#pragma unroll
            for (int o = 0; o < OUT_DIM; ++o) pre[o] += v[ii][o] * (ev[o] * inv);
        }
        if (it < 2) {
            float pn[OUT_DIM];
#pragma unroll
            for (int o = 0; o < OUT_DIM; ++o) pn[o] = sqrtf(reduce16(pre[o] * pre[o]));
#pragma unroll
            for (int ii = 0; ii < IN_DIM; ++ii)
#pragma unroll
                for (int o = 0; o < OUT_DIM; ++o) {
                    float d  = reduce16(pre[o] * v[ii][o]);
                    float w2 = reduce16(v[ii][o] * v[ii][o]);
                    float dist = d / fmaxf(pn[o] * sqrtf(w2), 1e-8f);
                    logits[ii][o] += dist;
                }
        } else {
            // squash over out_atoms and write activation
#pragma unroll
            for (int o = 0; o < OUT_DIM; ++o) {
                float s = reduce16(pre[o] * pre[o]);
                float n = sqrtf(s);
                float scale = (s / (1.0f + s)) / (n + 1e-12f);
                size_t idx = (((size_t)(b * OUT_DIM + o) * OUT_ATOMS + a) << 15)
                           + (size_t)(z * 1024 + y * 32 + x0 + m);
                out[idx] = pre[o] * scale;
            }
        }
    }
}

extern "C" void kernel_launch(void* const* d_in, const int* in_sizes, int n_in,
                              void* d_out, int out_size, void* d_ws, size_t ws_size,
                              hipStream_t stream) {
    (void)in_sizes; (void)n_in; (void)out_size; (void)ws_size;
    const float* x      = (const float*)d_in[0];
    const float* conv_w = (const float*)d_in[1];
    const float* conv_b = (const float*)d_in[2];
    const float* biases = (const float*)d_in[3];
    float* out = (float*)d_out;
    _Float16* bpack = (_Float16*)d_ws;   // 114688 B of scratch

    prep_weights_kernel<<<224, 256, 0, stream>>>(conv_w, bpack);
    caps_fused_kernel<<<4096, 256, 0, stream>>>(x, bpack, conv_b, biases, out);
}
